// MHA_CommBlock_66666482368901
// MI455X (gfx1250) — matrix operand order repaired and checked by reading, not compiled, not run on hardware
//
#include <hip/hip_runtime.h>

// ---------------------------------------------------------------------------
// MHA comm block for MI455X (gfx1250, wave32, WMMA + Tensor Data Mover).
// B=8, N=2048, IN=256, H=8, D=64, HD=512.
// qkv_proj (bf16 WMMA, V written pre-transposed) ->
// flash attention (TDM double-buffered K/V^T staging, bf16 WMMA, fp32 softmax) ->
// out_proj (bf16 WMMA, fp32 out).
// ---------------------------------------------------------------------------

#define DEV __device__ __forceinline__

typedef __attribute__((ext_vector_type(16))) __bf16 v16bf;
typedef __attribute__((ext_vector_type(8)))  __bf16 v8bf;
typedef __attribute__((ext_vector_type(8)))  float  v8f;
typedef __attribute__((ext_vector_type(4)))  unsigned int u32x4;
typedef __attribute__((ext_vector_type(8)))  int    i32x8;
typedef __attribute__((ext_vector_type(4)))  int    i32x4;

// fp32 -> bf16 via native hardware convert (v_cvt_*_bf16_f32 on gfx1250)
DEV unsigned short f2bf(float f) {
    __bf16 b = (__bf16)f;
    unsigned short u; __builtin_memcpy(&u, &b, 2);
    return u;
}

DEV v16bf join16(v8bf lo, v8bf hi) {
    return __builtin_shufflevector(lo, hi, 0,1,2,3,4,5,6,7,8,9,10,11,12,13,14,15);
}
DEV v8bf ld8u(const unsigned short* p) { return *reinterpret_cast<const v8bf*>(p); }

DEV v8f wmma_bf16(v16bf a, v16bf b, v8f c) {
    // (neg_a, A, neg_b, B, c_mod, C, reuse_a, reuse_b)
    return __builtin_amdgcn_wmma_f32_16x16x32_bf16(false, a, false, b, (short)0, c, false, false);
}

// A fragment (16x32 bf16, row-major source). base = &row[k0].
// lanes 0-15 hold K = {0..15}; lanes 16-31 hold K = {16..31}.
DEV v16bf load_afrag(const unsigned short* base, int lane) {
    int kb = ((lane >> 4) & 1) * 16;
    return join16(ld8u(base + kb), ld8u(base + kb + 8));
}
// B fragment (32x16 bf16): p = &mat[n][k0 + (lane>>4)*16], 16 contiguous K elements.
DEV v16bf load_bfrag(const unsigned short* p) {
    return join16(ld8u(p), ld8u(p + 8));
}

// ---------------------------------------------------------------------------
// Tensor Data Mover: 2D tile load global -> LDS.
// Tile: tile_rows rows x 32 DWORDs (128B) per row; global row stride
// row_stride_dw DWORDs; LDS destination padded +4 DWORDs per 32 DWORDs
// (matches an LDS layout of [rows][72] unsigned short).
// ---------------------------------------------------------------------------
DEV void tdm_load_2d_128B(unsigned lds_off, const void* gptr,
                          unsigned row_stride_dw, unsigned tile_rows) {
    unsigned long long ga = (unsigned long long)(uintptr_t)gptr;
    u32x4 g0;
    g0[0] = 1u;                                           // count=1 (valid D#)
    g0[1] = lds_off;                                      // lds_addr
    g0[2] = (unsigned)(ga & 0xffffffffu);                 // global_addr[31:0]
    g0[3] = (unsigned)((ga >> 32) & 0x1ffffffu)           // global_addr[56:32]
          | (2u << 30);                                   // type=2 ("image")
    i32x8 g1;
    g1[0] = (int)((2u << 16)       // data_size = 4 bytes
          | (1u << 20)             // pad_enable
          | (4u << 22)             // pad_interval: every 32 DWORDs
          | (3u << 25));           // pad_amount: 4 DWORDs (16B)
    g1[1] = (int)(32u << 16);      // tensor_dim0 = 32 DWORDs (low 16 @ [63:48])
    g1[2] = (int)(tile_rows << 16);// tensor_dim1 (low 16 @ [95:80])
    g1[3] = (int)(32u << 16);      // tile_dim0 = 32 DWORDs @ [127:112]
    g1[4] = (int)tile_rows;        // tile_dim1 @ [143:128]
    g1[5] = (int)row_stride_dw;    // tensor_dim0_stride low 32 @ [191:160]
    g1[6] = 0;
    g1[7] = 0;
    i32x4 gz = {0, 0, 0, 0};
#if defined(__clang_major__) && (__clang_major__ >= 23)
    i32x8 z8 = {0, 0, 0, 0, 0, 0, 0, 0};
    __builtin_amdgcn_tensor_load_to_lds(g0, g1, gz, gz, z8, 0);
#else
    __builtin_amdgcn_tensor_load_to_lds(g0, g1, gz, gz, 0);
#endif
}

// ---------------------------------------------------------------------------
// Kernel 1: Q/K/V projection, K=256.  grid (128, 8), block 256 (8 waves).
// vt_mode=0: dst[(b*8+h)][n][d]   (Q, K layout)
// vt_mode=1: dst[(b*8+h)][d][n]   (V stored pre-transposed for TDM staging)
// ---------------------------------------------------------------------------
__global__ __launch_bounds__(256) void qkv_proj_kernel(
    const float* __restrict__ msg, const float* __restrict__ W,
    const float* __restrict__ bias, unsigned short* __restrict__ dst,
    int vt_mode)
{
    __shared__ alignas(32) unsigned short Wt[64][264];  // W^T slice, bf16

    const int tid  = threadIdx.x, lane = tid & 31, wave = tid >> 5;
    const int c0   = blockIdx.y * 64;
    const int m0   = blockIdx.x * 128;

    for (int i = tid; i < 64 * 256; i += 256) {
        int k = i >> 6, c = i & 63;
        Wt[c][k] = f2bf(W[k * 512 + c0 + c]);
    }
    __syncthreads();

    const int    row  = m0 + wave * 16 + (lane & 15);
    const float* arow = msg + (size_t)row * 256;
    const int    kb   = ((lane >> 4) & 1) * 8;
    const int    hi16 = (lane >> 4) * 16;

    v8f acc[4] = {{0,0,0,0,0,0,0,0},{0,0,0,0,0,0,0,0},{0,0,0,0,0,0,0,0},{0,0,0,0,0,0,0,0}};

    #pragma unroll
    for (int k0 = 0; k0 < 256; k0 += 32) {
        v8f r1 = *reinterpret_cast<const v8f*>(arow + k0 + hi16);
        v8f r2 = *reinterpret_cast<const v8f*>(arow + k0 + hi16 + 8);
        v16bf a;
        #pragma unroll
        for (int j = 0; j < 8; ++j) { a[j] = (__bf16)r1[j]; a[8 + j] = (__bf16)r2[j]; }
        #pragma unroll
        for (int t = 0; t < 4; ++t) {
            const unsigned short* bp = &Wt[t * 16 + (lane & 15)][k0 + hi16];
            acc[t] = wmma_bf16(a, load_bfrag(bp), acc[t]);
        }
    }

    #pragma unroll
    for (int t = 0; t < 4; ++t) {
        int   c  = c0 + t * 16 + (lane & 15);
        float bv = bias[c];
        int   h  = c >> 6, d = c & 63;
        #pragma unroll
        for (int i = 0; i < 8; ++i) {
            int rg = m0 + wave * 16 + (lane >> 4) * 8 + i;
            int b  = rg >> 11, nseq = rg & 2047;
            size_t idx = vt_mode
                ? (((size_t)(b * 8 + h)) * 64 + d) * 2048 + nseq        // [bh][d][n]
                : (((size_t)(b * 8 + h)) * 2048 + nseq) * 64 + d;       // [bh][n][d]
            dst[idx] = f2bf(acc[t][i] + bv);
        }
    }
}

// ---------------------------------------------------------------------------
// Kernel 2: flash attention per (b,h).  grid (32, 64), block 128 (4 waves).
// K blocks and V^T blocks are DMA'd by the TDM into double-buffered LDS.
// ---------------------------------------------------------------------------
__global__ __launch_bounds__(128) void attn_kernel(
    const unsigned short* __restrict__ Qh, const unsigned short* __restrict__ Kh,
    const unsigned short* __restrict__ Vt_g, unsigned short* __restrict__ Ctx)
{
    __shared__ alignas(16) unsigned short Ks[2][64][72];   // K block   [key][d]
    __shared__ alignas(16) unsigned short Vs[2][64][72];   // V block^T [d][key]
    __shared__ alignas(32) unsigned short Pst[4][16][72];  // per-wave P staging

    const int tid  = threadIdx.x, lane = tid & 31, wave = tid >> 5;
    const int bh   = blockIdx.y;
    const int q0   = blockIdx.x * 64;
    const int hi16 = (lane >> 4) * 16;

    const size_t kvbase = (size_t)bh * 2048 * 64;   // elements per (b,h) plane
    const unsigned short* qrow = Qh + kvbase + (size_t)(q0 + wave * 16 + (lane & 15)) * 64;
    const v16bf qf0 = load_afrag(qrow, lane);
    const v16bf qf1 = load_afrag(qrow + 32, lane);

    // Prologue: TDM fetch of key-block 0 into buffer 0 (wave 0 only).
    if (wave == 0) {
        tdm_load_2d_128B((unsigned)(uintptr_t)&Ks[0][0][0], Kh + kvbase, 32, 64);
        tdm_load_2d_128B((unsigned)(uintptr_t)&Vs[0][0][0], Vt_g + kvbase, 1024, 64);
    }

    float m_run[8], l_run[8];
    v8f   o[4];
    #pragma unroll
    for (int i = 0; i < 8; ++i) { m_run[i] = -3.0e38f; l_run[i] = 0.0f; }
    #pragma unroll
    for (int t = 0; t < 4; ++t) { o[t] = (v8f){0,0,0,0,0,0,0,0}; }

    for (int kb0 = 0; kb0 < 2048; kb0 += 64) {
        const int buf = (kb0 >> 6) & 1;
        if (wave == 0) {
            if (kb0 + 64 < 2048) {   // prefetch next block into the other buffer
                const int nb = buf ^ 1;
                tdm_load_2d_128B((unsigned)(uintptr_t)&Ks[nb][0][0],
                                 Kh + kvbase + (size_t)(kb0 + 64) * 64, 32, 64);
                tdm_load_2d_128B((unsigned)(uintptr_t)&Vs[nb][0][0],
                                 Vt_g + kvbase + (size_t)(kb0 + 64), 1024, 64);
                __builtin_amdgcn_s_wait_tensorcnt(2);  // current block's 2 loads done
            } else {
                __builtin_amdgcn_s_wait_tensorcnt(0);
            }
        }
        __syncthreads();   // current block visible to all waves

        // S = Q K^T (four 16x16 tiles across 64 keys)
        v8f sc[4];
        #pragma unroll
        for (int t = 0; t < 4; ++t) {
            sc[t] = (v8f){0,0,0,0,0,0,0,0};
            const unsigned short* b0 = &Ks[buf][t * 16 + (lane & 15)][hi16];
            sc[t] = wmma_bf16(qf0, load_bfrag(b0), sc[t]);
            sc[t] = wmma_bf16(qf1, load_bfrag(b0 + 32), sc[t]);
        }

        // scale, exclude-self mask, online softmax (16-lane row reductions)
        const int qbase = q0 + wave * 16 + (lane >> 4) * 8;
        #pragma unroll
        for (int i = 0; i < 8; ++i) {
            float mx = -3.0e38f;
            #pragma unroll
            for (int t = 0; t < 4; ++t) {
                float s  = sc[t][i] * 0.125f;            // 1/sqrt(64)
                int   kg = kb0 + t * 16 + (lane & 15);
                if (qbase + i == kg) s = -1.0e9f;
                sc[t][i] = s;
                mx = fmaxf(mx, s);
            }
            #pragma unroll
            for (int off = 1; off < 16; off <<= 1) mx = fmaxf(mx, __shfl_xor(mx, off, 32));

            float nm   = fmaxf(m_run[i], mx);
            float resc = __expf(m_run[i] - nm);
            m_run[i]   = nm;
            float psum = 0.0f;
            #pragma unroll
            for (int t = 0; t < 4; ++t) {
                float p = __expf(sc[t][i] - nm);
                psum += p;
                Pst[wave][(lane >> 4) * 8 + i][t * 16 + (lane & 15)] = f2bf(p);
            }
            #pragma unroll
            for (int off = 1; off < 16; off <<= 1) psum += __shfl_xor(psum, off, 32);
            l_run[i] = l_run[i] * resc + psum;
            #pragma unroll
            for (int t = 0; t < 4; ++t) o[t][i] *= resc;
        }

        // O += P V  (P re-read from per-wave LDS strip in A layout; DS in-order)
        const unsigned short* prow = &Pst[wave][lane & 15][0];
        const v16bf pf0 = load_afrag(prow, lane);
        const v16bf pf1 = load_afrag(prow + 32, lane);
        #pragma unroll
        for (int t = 0; t < 4; ++t) {
            const unsigned short* b0 = &Vs[buf][t * 16 + (lane & 15)][hi16];
            o[t] = wmma_bf16(pf0, load_bfrag(b0), o[t]);
            o[t] = wmma_bf16(pf1, load_bfrag(b0 + 32), o[t]);
        }
        __syncthreads();   // all waves done with buf before TDM may overwrite it
    }

    // normalize and write ctx (bf16, [B][N][H*D] layout for the output GEMM)
    const int b = bh >> 3, h = bh & 7;
    #pragma unroll
    for (int i = 0; i < 8; ++i) {
        float inv = 1.0f / l_run[i];
        int   rg  = q0 + wave * 16 + (lane >> 4) * 8 + i;
        #pragma unroll
        for (int t = 0; t < 4; ++t) {
            int col = h * 64 + t * 16 + (lane & 15);
            Ctx[((size_t)b * 2048 + rg) * 512 + col] = f2bf(o[t][i] * inv);
        }
    }
}

// ---------------------------------------------------------------------------
// Kernel 3: out = ctx @ Wo (fp32 out).  grid (128), block 256 (8 waves).
// ---------------------------------------------------------------------------
__global__ __launch_bounds__(256) void out_proj_kernel(
    const unsigned short* __restrict__ Ctx, const float* __restrict__ Wo,
    float* __restrict__ out)
{
    __shared__ alignas(32) unsigned short Wot[64][264];  // Wo^T chunk, bf16

    const int tid  = threadIdx.x, lane = tid & 31, wave = tid >> 5;
    const int m0   = blockIdx.x * 128;
    const int hi16 = (lane >> 4) * 16;

    const unsigned short* arow = Ctx + (size_t)(m0 + wave * 16 + (lane & 15)) * 512;

    v8f acc[4] = {{0,0,0,0,0,0,0,0},{0,0,0,0,0,0,0,0},{0,0,0,0,0,0,0,0},{0,0,0,0,0,0,0,0}};

    for (int kc = 0; kc < 512; kc += 256) {
        __syncthreads();
        for (int i = tid; i < 64 * 256; i += 256) {
            int k = i >> 6, c = i & 63;
            Wot[c][k] = f2bf(Wo[(kc + k) * 64 + c]);
        }
        __syncthreads();

        #pragma unroll
        for (int k0 = 0; k0 < 256; k0 += 32) {
            v16bf a = load_afrag(arow + kc + k0, lane);
            #pragma unroll
            for (int t = 0; t < 4; ++t) {
                const unsigned short* bp = &Wot[t * 16 + (lane & 15)][k0 + hi16];
                acc[t] = wmma_bf16(a, load_bfrag(bp), acc[t]);
            }
        }
    }

    #pragma unroll
    for (int t = 0; t < 4; ++t) {
        #pragma unroll
        for (int i = 0; i < 8; ++i) {
            int rg = m0 + wave * 16 + (lane >> 4) * 8 + i;
            out[(size_t)rg * 64 + t * 16 + (lane & 15)] = acc[t][i];
        }
    }
}

// ---------------------------------------------------------------------------
extern "C" void kernel_launch(void* const* d_in, const int* in_sizes, int n_in,
                              void* d_out, int out_size, void* d_ws, size_t ws_size,
                              hipStream_t stream) {
    const float* msg = (const float*)d_in[0];
    const float* Wq  = (const float*)d_in[1];
    const float* bq  = (const float*)d_in[2];
    const float* Wk  = (const float*)d_in[3];
    const float* bk  = (const float*)d_in[4];
    const float* Wv  = (const float*)d_in[5];
    const float* bv  = (const float*)d_in[6];
    const float* Wo  = (const float*)d_in[7];

    unsigned short* ws = (unsigned short*)d_ws;
    const size_t seg = (size_t)64 * 2048 * 64;  // 8,388,608 bf16 elems = 16 MB
    unsigned short* Qh  = ws;
    unsigned short* Kh  = ws + seg;
    unsigned short* Vt  = ws + 2 * seg;   // pre-transposed: [bh][d][n]
    unsigned short* Ctx = ws + 3 * seg;

    dim3 gProj(128, 8);
    qkv_proj_kernel<<<gProj, 256, 0, stream>>>(msg, Wq, bq, Qh, 0);
    qkv_proj_kernel<<<gProj, 256, 0, stream>>>(msg, Wk, bk, Kh, 0);
    qkv_proj_kernel<<<gProj, 256, 0, stream>>>(msg, Wv, bv, Vt, 1);

    attn_kernel<<<dim3(32, 64), 128, 0, stream>>>(Qh, Kh, Vt, Ctx);

    out_proj_kernel<<<dim3(128), 256, 0, stream>>>(Ctx, Wo, (float*)d_out);
}
